// FrequencyRepresentationModule_2D_v0519_28802050687303
// MI455X (gfx1250) — matrix operand order repaired and checked by reading, not compiled, not run on hardware
//
#include <hip/hip_runtime.h>
#include <hip/hip_bf16.h>
#include <math.h>

// ---------------- problem constants ----------------
#define BB   32
#define AA   4
#define DD   128          // D0 == D1 == 128
#define NHH  6
#define OUTHW 513         // (128-1)*4 + 8 - 4 + 1

typedef __attribute__((ext_vector_type(16))) _Float16 v16h;
typedef __attribute__((ext_vector_type(8)))  _Float16 v8h;
typedef __attribute__((ext_vector_type(8)))  float    v8f;

// ---------------- WMMA helpers (CDNA5 / gfx1250, wave32) ----------------

__device__ __forceinline__ v8f wmma_f16(v16h a, v16h b, v8f c) {
  // D = A*B + C, f32 accumulate, emits v_wmma_f32_16x16x32_f16
  return __builtin_amdgcn_wmma_f32_16x16x32_f16(
      /*neg_a=*/false, a, /*neg_b=*/false, b,
      /*c_mod=*/(short)0, c, /*reuse_a=*/false, /*reuse_b=*/false);
}

// A-matrix 16x32 f16 per ISA layout:
//   lanes 0-15:  M=lane,    K = {k0+0..7} in v[0..7], {k0+16..23} in v[8..15]
//   lanes 16-31: M=lane-16, K = {k0+8..15},{k0+24..31}
__device__ __forceinline__ v16h load_a(const _Float16* __restrict__ base,
                                       int m0, int ld, int k0, int lane) {
  int r    = m0 + (lane & 15);
  int koff = (lane >> 4) << 3;
  const _Float16* p = base + r * ld + k0 + koff;
  v16h v;
#pragma unroll
  for (int i = 0; i < 8; ++i) { v[i] = p[i]; v[i + 8] = p[i + 16]; }
  return v;
}

// B-matrix 32x16 f16 per ISA layout (storage is N-major: base[n*ld + k]):
//   lanes 0-15:  N=lane,    K = {k0+0..7} in v[0..7], {k0+16..23} in v[8..15]
//   lanes 16-31: N=lane-16, K = {k0+8..15},{k0+24..31}
__device__ __forceinline__ v16h load_b(const _Float16* __restrict__ base,
                                       int n0, int ld, int k0, int lane) {
  int n    = n0 + (lane & 15);
  int koff = (lane >> 4) << 3;
  const _Float16* p = base + n * ld + k0 + koff;
  v16h v;
#pragma unroll
  for (int i = 0; i < 8; ++i) { v[i] = p[i]; v[i + 8] = p[i + 16]; }
  return v;
}

// C/D 16x16 f32: lane<16 -> N=lane, VGPR r -> M=r; lane>=16 -> N=lane-16, M=r+8
__device__ __forceinline__ void store_tile_f32(float* __restrict__ base,
                                               int m0, int n0, int ld,
                                               const v8f& c, int lane) {
  int n  = n0 + (lane & 15);
  int mb = m0 + ((lane >> 4) << 3);
#pragma unroll
  for (int r = 0; r < 8; ++r) base[(mb + r) * ld + n] = c[r];
}

// transposed f16 store: element (m,n) -> base[n*ld + m]; per-lane 8 halves contiguous
__device__ __forceinline__ void store_tile_f16_T(_Float16* __restrict__ base,
                                                 int m0, int n0, int ld,
                                                 const v8f& c, int lane) {
  int n  = n0 + (lane & 15);
  int mb = m0 + ((lane >> 4) << 3);
  v8h h;
#pragma unroll
  for (int r = 0; r < 8; ++r) h[r] = (_Float16)c[r];
  *(v8h*)(base + n * ld + mb) = h;
}

// complex GEMM core, one 16-row strip x four 16-col tiles (16x64 of C):
//   accR[t] += Wr*Br - Wi*Bi   (Wi negated in-register: packed f16 sign flip)
//   accI[t] += Wr*Bi + Wi*Br
// A: row-major MxK (ld=K), B: N-major NxK (ld=K).
// A fragments are loaded once per K-step and reused across the 4 N-tiles
// (amortizes the weight-matrix traffic; 10 loads -> 16 WMMAs per K-step).
__device__ __forceinline__ void cgemm_core4(const _Float16* __restrict__ Ar,
                                            const _Float16* __restrict__ Ai, int m0,
                                            const _Float16* __restrict__ Br,
                                            const _Float16* __restrict__ Bi, int n0,
                                            int K, int lane,
                                            v8f accR[4], v8f accI[4]) {
#pragma unroll
  for (int t = 0; t < 4; ++t) {
    accR[t] = (v8f){0,0,0,0,0,0,0,0};
    accI[t] = (v8f){0,0,0,0,0,0,0,0};
  }
  for (int k0 = 0; k0 < K; k0 += 32) {
    v16h ar = load_a(Ar, m0, K, k0, lane);
    v16h ai = load_a(Ai, m0, K, k0, lane);
    v16h an;
#pragma unroll
    for (int i = 0; i < 16; ++i) an[i] = -ai[i];
#pragma unroll
    for (int t = 0; t < 4; ++t) {
      v16h br = load_b(Br, n0 + (t << 4), K, k0, lane);
      v16h bi = load_b(Bi, n0 + (t << 4), K, k0, lane);
      accR[t] = wmma_f16(ar, br, accR[t]);
      accR[t] = wmma_f16(an, bi, accR[t]);
      accI[t] = wmma_f16(ar, bi, accI[t]);
      accI[t] = wmma_f16(ai, br, accI[t]);
    }
  }
}

// ---------------- prep kernels ----------------

__global__ void k_f32_to_f16(const float* __restrict__ in, _Float16* __restrict__ out, int n) {
  int i = blockIdx.x * blockDim.x + threadIdx.x;
  if (i < n) out[i] = (_Float16)in[i];
}

// inp (B,2,128,128) -> xh0r/xh0i (B,128,128) f16 (row-major (h,w) == N-major for k=w)
__global__ void k_split_inp(const float* __restrict__ inp,
                            _Float16* __restrict__ xr, _Float16* __restrict__ xi) {
  int i = blockIdx.x * blockDim.x + threadIdx.x;
  if (i >= BB * 16384) return;
  int b = i >> 14, j = i & 16383;
  xr[i] = (_Float16)inp[b * 32768 + j];
  xi[i] = (_Float16)inp[b * 32768 + 16384 + j];
}

// nmat x 128 x 128 f32 -> transposed f16
__global__ void k_transpose_f16(const float* __restrict__ in, _Float16* __restrict__ out, int n) {
  int i = blockIdx.x * blockDim.x + threadIdx.x;
  if (i >= n) return;
  int m = i >> 14, rc = i & 16383, r = rc >> 7, c = rc & 127;
  out[(m << 14) + (c << 7) + r] = (_Float16)in[i];
}

// ---------------- GEMM kernels (16x64 strip per wave) ----------------

// ar/ai = W1 (512x128) complex @ X^T per batch -> xreal/ximag (B,512,128) f32
__global__ void __launch_bounds__(32) k_gemm_w1(const _Float16* __restrict__ Ar,
                                                const _Float16* __restrict__ Ai,
                                                const _Float16* __restrict__ Br,
                                                const _Float16* __restrict__ Bi,
                                                float* __restrict__ Cr,
                                                float* __restrict__ Ci) {
  int b = blockIdx.x, m0 = blockIdx.y << 4, n0 = blockIdx.z << 6, lane = threadIdx.x;
  v8f aR[4], aI[4];
  cgemm_core4(Ar, Ai, m0, Br + (b << 14), Bi + (b << 14), n0, 128, lane, aR, aI);
#pragma unroll
  for (int t = 0; t < 4; ++t) {
    store_tile_f32(Cr + b * 65536, m0, n0 + (t << 4), 128, aR[t], lane);
    store_tile_f32(Ci + b * 65536, m0, n0 + (t << 4), 128, aI[t], lane);
  }
}

// cbranch stage 1: I = W @ x (complex), fuse f=1+log|I|^2, E=(f*Ir, -f*Ii) -> f16 transposed
__global__ void __launch_bounds__(32) k_cbranch_m1(const _Float16* __restrict__ Wr,
                                                   const _Float16* __restrict__ Wi,
                                                   const _Float16* __restrict__ Xr,
                                                   const _Float16* __restrict__ Xi,
                                                   _Float16* __restrict__ Er,
                                                   _Float16* __restrict__ Ei) {
  int b = blockIdx.x, m0 = blockIdx.y << 4, n0 = blockIdx.z << 6, lane = threadIdx.x;
  v8f aR[4], aI[4];
  cgemm_core4(Wr, Wi, m0, Xr + (b << 14), Xi + (b << 14), n0, 128, lane, aR, aI);
#pragma unroll
  for (int t = 0; t < 4; ++t) {
    v8f er, ei;
#pragma unroll
    for (int r = 0; r < 8; ++r) {
      float a = aR[t][r], c = aI[t][r];
      float f = 1.0f + logf(a * a + c * c);
      er[r] = f * a;
      ei[r] = -(f * c);
    }
    store_tile_f16_T(Er + (b << 14), m0, n0 + (t << 4), 128, er, lane);
    store_tile_f16_T(Ei + (b << 14), m0, n0 + (t << 4), 128, ei, lane);
  }
}

// cbranch stage 2: y = W^T @ E (complex) -> yr/yi (B,128,128) f32 row-major
__global__ void __launch_bounds__(32) k_cbranch_m2(const _Float16* __restrict__ WTr,
                                                   const _Float16* __restrict__ WTi,
                                                   const _Float16* __restrict__ Er,
                                                   const _Float16* __restrict__ Ei,
                                                   float* __restrict__ Yr,
                                                   float* __restrict__ Yi) {
  int b = blockIdx.x, m0 = blockIdx.y << 4, n0 = blockIdx.z << 6, lane = threadIdx.x;
  v8f aR[4], aI[4];
  cgemm_core4(WTr, WTi, m0, Er + (b << 14), Ei + (b << 14), n0, 128, lane, aR, aI);
#pragma unroll
  for (int t = 0; t < 4; ++t) {
    store_tile_f32(Yr + (b << 14), m0, n0 + (t << 4), 128, aR[t], lane);
    store_tile_f32(Yi + (b << 14), m0, n0 + (t << 4), 128, aI[t], lane);
  }
}

// f = W3 complex @ x, fuse magnitude -> mag (B,512,128) f32
__global__ void __launch_bounds__(32) k_gemm_w3(const _Float16* __restrict__ Ar,
                                                const _Float16* __restrict__ Ai,
                                                const _Float16* __restrict__ Xr,
                                                const _Float16* __restrict__ Xi,
                                                float* __restrict__ Mag) {
  int b = blockIdx.x, m0 = blockIdx.y << 4, n0 = blockIdx.z << 6, lane = threadIdx.x;
  v8f aR[4], aI[4];
  cgemm_core4(Ar, Ai, m0, Xr + (b << 14), Xi + (b << 14), n0, 128, lane, aR, aI);
#pragma unroll
  for (int t = 0; t < 4; ++t) {
    v8f mg;
#pragma unroll
    for (int r = 0; r < 8; ++r)
      mg[r] = sqrtf(aR[t][r] * aR[t][r] + aI[t][r] * aI[t][r]);
    store_tile_f32(Mag + b * 65536, m0, n0 + (t << 4), 128, mg, lane);
  }
}

// ---------------- phase-modulation stage ----------------

// cr/ci[b,o,w] = sum_{c,h} w2[o,c,h] * n[b,c,h,w]  (complex), w in [0,127)
__global__ void k_phase_c(const float* __restrict__ Xr, const float* __restrict__ Xi,
                          const float* __restrict__ w2r, const float* __restrict__ w2i,
                          float* __restrict__ cr, float* __restrict__ ci) {
  __shared__ float wr_s[512], wi_s[512];
  int o = blockIdx.x & 3, b = blockIdx.x >> 2;
  for (int j = threadIdx.x; j < 512; j += 128) {
    wr_s[j] = w2r[o * 512 + j];
    wi_s[j] = w2i[o * 512 + j];
  }
  __syncthreads();
  int w = threadIdx.x;
  if (w >= 127) return;
  float ar = 0.f, ai = 0.f;
  for (int c = 0; c < 4; ++c)
    for (int h = 0; h < 128; ++h) {
      const float* pr = Xr + (((b << 2) + c) << 14) + (h << 7) + w;
      const float* pi = Xi + (((b << 2) + c) << 14) + (h << 7) + w;
      float x0 = pr[0], x1 = pr[1], y0 = pi[0], y1 = pi[1];
      float nr = x0 * x1 + y0 * y1;
      float ni = x0 * y1 - y0 * x1;
      float wr = wr_s[c * 128 + h], wi = wi_s[c * 128 + h];
      ar += wr * nr - wi * ni;
      ai += wr * ni + wi * nr;
    }
  cr[blockIdx.x * 128 + w] = ar;
  ci[blockIdx.x * 128 + w] = ai;
}

// ang = cumsum(pad(atan2(ci,cr))), one (b,o) per block (serial scan, tiny)
__global__ void k_cumsum(const float* __restrict__ cr, const float* __restrict__ ci,
                         float* __restrict__ ang) {
  if (threadIdx.x != 0) return;
  int bo = blockIdx.x;
  float acc = 0.f;
  ang[bo * 128] = 0.f;
  for (int w = 0; w < 127; ++w) {
    acc += atan2f(ci[bo * 128 + w], cr[bo * 128 + w]);
    ang[bo * 128 + w + 1] = acc;
  }
}

// rotate xcp by ang, contract over a with wcp -> G (f32) and transposed f16 activations
__global__ void k_modulate(const float* __restrict__ Xr, const float* __restrict__ Xi,
                           const float* __restrict__ ang,
                           const float* __restrict__ wcpr, const float* __restrict__ wcpi,
                           float* __restrict__ Gr, float* __restrict__ Gi,
                           _Float16* __restrict__ xhr, _Float16* __restrict__ xhi) {
  int i = blockIdx.x * blockDim.x + threadIdx.x;
  if (i >= BB * 16384) return;
  int b = i >> 14, p = (i >> 7) & 127, q = i & 127;
  float gr = 0.f, gi = 0.f;
#pragma unroll
  for (int a = 0; a < 4; ++a) {
    float an = ang[((b << 2) + a) * 128 + p];
    float c = cosf(an), s = sinf(an);
    int idx = (((b << 2) + a) << 14) + (q << 7) + p;  // xreal[b,a,q,p] (transpose)
    float xr_ = Xr[idx], xi_ = Xi[idx];
    float rr = xr_ * c + xi_ * s;
    float ri = xi_ * c - xr_ * s;
    gr += wcpr[a] * rr - wcpi[a] * ri;
    gi += wcpr[a] * ri + wcpi[a] * rr;
  }
  Gr[i] = gr;
  Gi[i] = gi;
  // transposed (N-major) f16 copy for WMMA B operand: xh[b, w*128 + h]
  xhr[(b << 14) + (q << 7) + p] = (_Float16)gr;
  xhi[(b << 14) + (q << 7) + p] = (_Float16)gi;
}

// ---------------- cbranch tail + combine ----------------

// per (b,row): wr = sum_w Gr*yr - Gi*yi ; wi = sum_w Gr*yi + Gi*yr ; -> cos/sin(atan2)
__global__ void k_branch_reduce(const float* __restrict__ Gr, const float* __restrict__ Gi,
                                const float* __restrict__ Yr, const float* __restrict__ Yi,
                                float* __restrict__ cc, float* __restrict__ ss) {
  __shared__ float sr[128], si[128];
  int b = blockIdx.x >> 7, row = blockIdx.x & 127, t = threadIdx.x;
  int idx = (b << 14) + (row << 7) + t;
  float gr = Gr[idx], gi = Gi[idx], vr = Yr[idx], vi = Yi[idx];
  sr[t] = gr * vr - gi * vi;
  si[t] = gr * vi + gi * vr;
  __syncthreads();
  for (int s = 64; s > 0; s >>= 1) {
    if (t < s) { sr[t] += sr[t + s]; si[t] += si[t + s]; }
    __syncthreads();
  }
  if (t == 0) {
    float an = atan2f(si[0], sr[0]);
    cc[blockIdx.x] = cosf(an);
    ss[blockIdx.x] = sinf(an);
  }
}

// x_new = wc combination of the two rotated-G branches; write transposed f16 acts
__global__ void k_combine(const float* __restrict__ Gr, const float* __restrict__ Gi,
                          const float* __restrict__ c1, const float* __restrict__ s1,
                          const float* __restrict__ c2, const float* __restrict__ s2,
                          const float* __restrict__ wc_r, const float* __restrict__ wc_i,
                          int iter, _Float16* __restrict__ xhr, _Float16* __restrict__ xhi) {
  int i = blockIdx.x * blockDim.x + threadIdx.x;
  if (i >= BB * 16384) return;
  int b = i >> 14, k = (i >> 7) & 127, w = i & 127;
  float gr = Gr[i], gi = Gi[i];
  float C1 = c1[(b << 7) + k], S1 = s1[(b << 7) + k];
  float C2 = c2[(b << 7) + k], S2 = s2[(b << 7) + k];
  float r1 = gr * C1 + gi * S1, i1 = gi * C1 - gr * S1;
  float r2 = gr * C2 + gi * S2, i2 = gi * C2 - gr * S2;
  float wr0 = wc_r[iter * 2], wr1 = wc_r[iter * 2 + 1];
  float wi0 = wc_i[iter * 2], wi1 = wc_i[iter * 2 + 1];
  float xr = wr0 * r1 + wr1 * r2 - wi0 * i1 - wi1 * i2;
  float xi = wr0 * i1 + wr1 * i2 + wi0 * r1 + wi1 * r2;
  xhr[(b << 14) + (w << 7) + k] = (_Float16)xr;   // transposed for next GEMM
  xhi[(b << 14) + (w << 7) + k] = (_Float16)xi;
}

// ---------------- normalization + conv blocks + deconv ----------------

__global__ void k_batch_max(const float* __restrict__ mag, float* __restrict__ mx) {
  __shared__ float sm[256];
  int b = blockIdx.x, t = threadIdx.x;
  float m = 0.f;
  for (int j = t; j < 65536; j += 256) m = fmaxf(m, mag[b * 65536 + j]);
  sm[t] = m;
  __syncthreads();
  for (int s = 128; s > 0; s >>= 1) {
    if (t < s) sm[t] = fmaxf(sm[t], sm[t + s]);
    __syncthreads();
  }
  if (t == 0) mx[b] = sm[0];
}

__global__ void k_scale(const float* __restrict__ mag, const float* __restrict__ mx,
                        float* __restrict__ xc) {
  int i = blockIdx.x * blockDim.x + threadIdx.x;
  if (i >= BB * 65536) return;
  xc[i] = mag[i] / mx[i >> 16];
}

// 3x3 circular conv (4->4 ch) + BN (+optional ReLU)
__global__ void k_conv_bn(const float* __restrict__ xin, const float* __restrict__ wconv,
                          const float* __restrict__ g, const float* __restrict__ bb,
                          const float* __restrict__ mn, const float* __restrict__ vr,
                          float* __restrict__ out, int relu) {
  int i = blockIdx.x * blockDim.x + threadIdx.x;
  if (i >= BB * 4 * 16384) return;
  int x = i & 127, y = (i >> 7) & 127, o = (i >> 14) & 3, b = i >> 16;
  float acc = 0.f;
#pragma unroll
  for (int c = 0; c < 4; ++c)
#pragma unroll
    for (int ky = 0; ky < 3; ++ky) {
      int yy = (y + ky + 127) & 127;
#pragma unroll
      for (int kx = 0; kx < 3; ++kx) {
        int xx = (x + kx + 127) & 127;
        acc += xin[(((b << 2) + c) << 14) + (yy << 7) + xx] *
               wconv[(o * 4 + c) * 9 + ky * 3 + kx];
      }
    }
  float inv = g[o] * rsqrtf(vr[o] + 1e-5f);
  float r = (acc - mn[o]) * inv + bb[o];
  if (relu) r = fmaxf(r, 0.f);
  out[i] = r;
}

__global__ void k_resid(float* __restrict__ x, const float* __restrict__ h, int n) {
  int i = blockIdx.x * blockDim.x + threadIdx.x;
  if (i < n) x[i] = fmaxf(x[i] + h[i], 0.f);
}

// ConvTranspose2d(4->1, k=8, stride=4, pad=2, outpad=1) -> (B,513,513)
__global__ void k_deconv(const float* __restrict__ xc, const float* __restrict__ wout,
                         float* __restrict__ out) {
  int i = blockIdx.x * blockDim.x + threadIdx.x;
  if (i >= BB * OUTHW * OUTHW) return;
  int ox = i % OUTHW;
  int t = i / OUTHW;
  int oy = t % OUTHW;
  int b = t / OUTHW;
  float acc = 0.f;
#pragma unroll
  for (int ky = 0; ky < 8; ++ky) {
    int ty = oy + 2 - ky;
    if (ty < 0 || (ty & 3)) continue;
    int iy = ty >> 2;
    if (iy >= 128) continue;
#pragma unroll
    for (int kx = 0; kx < 8; ++kx) {
      int tx = ox + 2 - kx;
      if (tx < 0 || (tx & 3)) continue;
      int ix = tx >> 2;
      if (ix >= 128) continue;
#pragma unroll
      for (int c = 0; c < 4; ++c)
        acc += xc[(((b << 2) + c) << 14) + (iy << 7) + ix] * wout[c * 64 + ky * 8 + kx];
    }
  }
  out[i] = acc;
}

// ---------------- orchestration ----------------

extern "C" void kernel_launch(void* const* d_in, const int* in_sizes, int n_in,
                              void* d_out, int out_size, void* d_ws, size_t ws_size,
                              hipStream_t stream) {
  (void)in_sizes; (void)n_in; (void)out_size; (void)ws_size;
  const float* inp   = (const float*)d_in[0];
  const float* w1_r  = (const float*)d_in[1];
  const float* w1_i  = (const float*)d_in[2];
  const float* w2_r  = (const float*)d_in[3];
  const float* w2_i  = (const float*)d_in[4];
  const float* wcp_r = (const float*)d_in[5];
  const float* wcp_i = (const float*)d_in[6];
  const float* wm_r  = (const float*)d_in[7];
  const float* wm_i  = (const float*)d_in[8];
  const float* wc_r  = (const float*)d_in[9];
  const float* wc_i  = (const float*)d_in[10];
  const float* w3_r  = (const float*)d_in[11];
  const float* w3_i  = (const float*)d_in[12];
  const float* m2_w1 = (const float*)d_in[13];
  const float* m2_w2 = (const float*)d_in[14];
  const float* bn1_g = (const float*)d_in[15];
  const float* bn1_b = (const float*)d_in[16];
  const float* bn1_m = (const float*)d_in[17];
  const float* bn1_v = (const float*)d_in[18];
  const float* bn2_g = (const float*)d_in[19];
  const float* bn2_b = (const float*)d_in[20];
  const float* bn2_m = (const float*)d_in[21];
  const float* bn2_v = (const float*)d_in[22];
  const float* wout  = (const float*)d_in[23];
  float* out = (float*)d_out;

  char* wsb = (char*)d_ws;
  size_t off = 0;
  auto alloc = [&](size_t bytes) -> void* {
    void* p = wsb + off;
    off += (bytes + 255) & ~(size_t)255;
    return p;
  };

  const size_t HB = sizeof(_Float16);
  _Float16* xh0r  = (_Float16*)alloc((size_t)BB * 16384 * HB);
  _Float16* xh0i  = (_Float16*)alloc((size_t)BB * 16384 * HB);
  _Float16* w1hr  = (_Float16*)alloc(65536 * HB);
  _Float16* w1hi  = (_Float16*)alloc(65536 * HB);
  _Float16* w3hr  = (_Float16*)alloc(65536 * HB);
  _Float16* w3hi  = (_Float16*)alloc(65536 * HB);
  _Float16* wmhr  = (_Float16*)alloc(12 * 16384 * HB);
  _Float16* wmhi  = (_Float16*)alloc(12 * 16384 * HB);
  _Float16* wmThr = (_Float16*)alloc(12 * 16384 * HB);
  _Float16* wmThi = (_Float16*)alloc(12 * 16384 * HB);
  float* xreal = (float*)alloc((size_t)BB * 65536 * 4);
  float* ximag = (float*)alloc((size_t)BB * 65536 * 4);
  float* crb   = (float*)alloc((size_t)BB * 4 * 128 * 4);
  float* cib   = (float*)alloc((size_t)BB * 4 * 128 * 4);
  float* ang   = (float*)alloc((size_t)BB * 4 * 128 * 4);
  float* Gr    = (float*)alloc((size_t)BB * 16384 * 4);
  float* Gi    = (float*)alloc((size_t)BB * 16384 * 4);
  _Float16* xhr = (_Float16*)alloc((size_t)BB * 16384 * HB);
  _Float16* xhi = (_Float16*)alloc((size_t)BB * 16384 * HB);
  _Float16* Ehr = (_Float16*)alloc((size_t)BB * 16384 * HB);
  _Float16* Ehi = (_Float16*)alloc((size_t)BB * 16384 * HB);
  float* yr = (float*)alloc((size_t)BB * 16384 * 4);
  float* yi = (float*)alloc((size_t)BB * 16384 * 4);
  float* c1 = (float*)alloc((size_t)BB * 128 * 4);
  float* s1 = (float*)alloc((size_t)BB * 128 * 4);
  float* c2 = (float*)alloc((size_t)BB * 128 * 4);
  float* s2 = (float*)alloc((size_t)BB * 128 * 4);
  float* mag = (float*)alloc((size_t)BB * 65536 * 4);
  float* mx  = (float*)alloc(BB * 4);
  float* xc  = (float*)alloc((size_t)BB * 65536 * 4);
  float* h1  = (float*)alloc((size_t)BB * 65536 * 4);
  float* h2  = (float*)alloc((size_t)BB * 65536 * 4);

  // ---- prep: f16 conversions ----
  k_split_inp<<<2048, 256, 0, stream>>>(inp, xh0r, xh0i);
  k_f32_to_f16<<<256, 256, 0, stream>>>(w1_r, w1hr, 65536);
  k_f32_to_f16<<<256, 256, 0, stream>>>(w1_i, w1hi, 65536);
  k_f32_to_f16<<<256, 256, 0, stream>>>(w3_r, w3hr, 65536);
  k_f32_to_f16<<<256, 256, 0, stream>>>(w3_i, w3hi, 65536);
  k_f32_to_f16<<<768, 256, 0, stream>>>(wm_r, wmhr, 12 * 16384);
  k_f32_to_f16<<<768, 256, 0, stream>>>(wm_i, wmhi, 12 * 16384);
  k_transpose_f16<<<768, 256, 0, stream>>>(wm_r, wmThr, 12 * 16384);
  k_transpose_f16<<<768, 256, 0, stream>>>(wm_i, wmThi, 12 * 16384);

  // ---- stage 1: w1 complex GEMM -> xreal/ximag (B,512,128) ----
  k_gemm_w1<<<dim3(BB, 32, 2), 32, 0, stream>>>(w1hr, w1hi, xh0r, xh0i, xreal, ximag);

  // ---- stage 2: phase-difference features, cumsum angle, modulate, contract ----
  k_phase_c<<<BB * 4, 128, 0, stream>>>(xreal, ximag, w2_r, w2_i, crb, cib);
  k_cumsum<<<BB * 4, 32, 0, stream>>>(crb, cib, ang);
  k_modulate<<<2048, 256, 0, stream>>>(xreal, ximag, ang, wcp_r, wcp_i, Gr, Gi, xhr, xhi);

  // ---- stage 3: 6 iterations of dual complex branches ----
  for (int i = 0; i < NHH; ++i) {
    int j = i + NHH;
    k_cbranch_m1<<<dim3(BB, 8, 2), 32, 0, stream>>>(wmhr + i * 16384, wmhi + i * 16384,
                                                    xhr, xhi, Ehr, Ehi);
    k_cbranch_m2<<<dim3(BB, 8, 2), 32, 0, stream>>>(wmThr + i * 16384, wmThi + i * 16384,
                                                    Ehr, Ehi, yr, yi);
    k_branch_reduce<<<BB * 128, 128, 0, stream>>>(Gr, Gi, yr, yi, c1, s1);

    k_cbranch_m1<<<dim3(BB, 8, 2), 32, 0, stream>>>(wmhr + j * 16384, wmhi + j * 16384,
                                                    xhr, xhi, Ehr, Ehi);
    k_cbranch_m2<<<dim3(BB, 8, 2), 32, 0, stream>>>(wmThr + j * 16384, wmThi + j * 16384,
                                                    Ehr, Ehi, yr, yi);
    k_branch_reduce<<<BB * 128, 128, 0, stream>>>(Gr, Gi, yr, yi, c2, s2);

    k_combine<<<2048, 256, 0, stream>>>(Gr, Gi, c1, s1, c2, s2, wc_r, wc_i, i, xhr, xhi);
  }

  // ---- stage 4: w3 complex GEMM + magnitude, batch-max normalize ----
  k_gemm_w3<<<dim3(BB, 32, 2), 32, 0, stream>>>(w3hr, w3hi, xhr, xhi, mag);
  k_batch_max<<<BB, 256, 0, stream>>>(mag, mx);
  k_scale<<<8192, 256, 0, stream>>>(mag, mx, xc);

  // ---- stage 5: residual circular-conv blocks ----
  for (int i = 0; i < NHH; ++i) {
    k_conv_bn<<<8192, 256, 0, stream>>>(xc, m2_w1 + i * 144,
                                        bn1_g + i * 4, bn1_b + i * 4, bn1_m + i * 4, bn1_v + i * 4,
                                        h1, 1);
    k_conv_bn<<<8192, 256, 0, stream>>>(h1, m2_w2 + i * 144,
                                        bn2_g + i * 4, bn2_b + i * 4, bn2_m + i * 4, bn2_v + i * 4,
                                        h2, 0);
    k_resid<<<8192, 256, 0, stream>>>(xc, h2, BB * 65536);
  }

  // ---- stage 6: stride-4 transposed conv -> (B,513,513) ----
  int nOut = BB * OUTHW * OUTHW;
  k_deconv<<<(nOut + 255) / 256, 256, 0, stream>>>(xc, wout, out);
}
